// Model_39676907887667
// MI455X (gfx1250) — matrix operand order repaired and checked by reading, not compiled, not run on hardware
//
#include <hip/hip_runtime.h>

typedef __attribute__((ext_vector_type(16))) _Float16 v16h;
typedef __attribute__((ext_vector_type(8)))  _Float16 v8h;
typedef __attribute__((ext_vector_type(8)))  float    v8f;

#define B_ 4
#define S_ 4096
#define D_ 64
#define NBLK (S_ / 64)

#if defined(__AMDGCN__) && __has_builtin(__builtin_amdgcn_tensor_load_to_lds) && \
    __has_builtin(__builtin_amdgcn_s_wait_tensorcnt)
#define USE_TDM 1
typedef __attribute__((ext_vector_type(4))) unsigned int tdm_u32x4;
typedef __attribute__((ext_vector_type(8))) int          tdm_i32x8;
typedef __attribute__((ext_vector_type(4))) int          tdm_i32x4;
#endif

// ---------------------------------------------------------------------------
// Kernel 1: fused projections.
//   qwd = qw * (q @ Wq^T + bq) / D      (f16)
//   kwd = kw * (k @ Wk^T + bk) / D      (f16)
//   vp  =      (v @ Wv^T + bv)          (f16)
// ---------------------------------------------------------------------------
__global__ __launch_bounds__(256) void proj_kernel(
    const float* __restrict__ q, const float* __restrict__ k, const float* __restrict__ v,
    const float* __restrict__ Wq, const float* __restrict__ bq,
    const float* __restrict__ Wk, const float* __restrict__ bk,
    const float* __restrict__ Wv, const float* __restrict__ bv,
    const float* __restrict__ qw, const float* __restrict__ kw,
    _Float16* __restrict__ qwd, _Float16* __restrict__ kwd, _Float16* __restrict__ vpo)
{
    __shared__ float qs[4 * 64], ks[4 * 64], vs[4 * 64];
    const int tid = threadIdx.x;
    const long rowbase = (long)blockIdx.x * 4;

    qs[tid] = q[rowbase * 64 + tid];
    ks[tid] = k[rowbase * 64 + tid];
    vs[tid] = v[rowbase * 64 + tid];
    __syncthreads();

    const int lr = tid >> 6;
    const int dd = tid & 63;
    const float* qrow = qs + lr * 64;
    const float* krow = ks + lr * 64;
    const float* vrow = vs + lr * 64;
    const float* wqr = Wq + dd * 64;
    const float* wkr = Wk + dd * 64;
    const float* wvr = Wv + dd * 64;

    float aq = bq[dd], ak = bk[dd], av = bv[dd];
    #pragma unroll 8
    for (int i = 0; i < 64; ++i) {
        aq = fmaf(qrow[i], wqr[i], aq);
        ak = fmaf(krow[i], wkr[i], ak);
        av = fmaf(vrow[i], wvr[i], av);
    }
    const long o = (rowbase + lr) * 64 + dd;
    qwd[o] = (_Float16)(qw[dd] * aq * (1.0f / 64.0f));
    kwd[o] = (_Float16)(kw[dd] * ak * (1.0f / 64.0f));
    vpo[o] = (_Float16)av;
}

// ---------------------------------------------------------------------------
// Kernel 2: flash attention. 8 waves/block, wave owns a 16-query tile.
// 64 tokens per iteration -> 16 v_wmma_f32_16x16x32_f16 per iteration.
// Double-buffered staging: TDM K-load + V transpose for block i+1 overlap the
// compute of block i; one barrier per iteration; tensorcnt fenced at loop top.
// ---------------------------------------------------------------------------
__global__ __launch_bounds__(256) void fa_kernel(
    const _Float16* __restrict__ qwd, const _Float16* __restrict__ kwd,
    const _Float16* __restrict__ vp, const float* __restrict__ mask,
    float* __restrict__ out)
{
    __shared__ _Float16 Klds[2 * 64 * 64];   // [buf][token][feat]
    __shared__ _Float16 Vt[2 * 64 * 64];     // [buf][feat][token] (transposed V)
    __shared__ _Float16 Pl[8 * 16 * 64];     // per-wave P scratch: [row][token]

    const int tid  = threadIdx.x;
    const int lane = tid & 31;
    const int wv   = tid >> 5;
    const int b    = blockIdx.y;
    const long qrow0 = (long)b * S_ + (long)blockIdx.x * 128 + wv * 16;
    const float mval = mask[0];

    const int nn = lane & 15;    // column / A-row selector
    const int hi = lane >> 4;    // lane-half

    // ---- Stage key/value block `blk` into LDS buffer `bf` (async-ish) ----
    auto stage = [&](int blk, int bf) {
        const _Float16* kg = kwd + ((long)b * S_ + blk * 64) * 64;
        _Float16* kd = Klds + bf * 4096;
#ifdef USE_TDM
        if (wv == 0) {
            // 2D tensor DMA descriptor (D#): 64x64 tile of 2-byte elements.
            unsigned long long ga = (unsigned long long)kg;
            unsigned ldsa = (unsigned)(unsigned long long)(void*)kd;
            tdm_u32x4 g0;
            g0[0] = 1u;                                  // count=1 (valid user D#)
            g0[1] = ldsa;                                // lds_addr
            g0[2] = (unsigned)(ga & 0xffffffffu);        // global_addr[31:0]
            g0[3] = (unsigned)((ga >> 32) & 0x1ffffffu)  // global_addr[56:32]
                    | (2u << 30);                        // type=2 ("image")
            tdm_i32x8 g1;
            g1[0] = (1 << 16);                           // data_size=2B, wg_mask=0
            g1[1] = (64 << 16);                          // tensor_dim0 = 64
            g1[2] = (64 << 16);                          // tensor_dim1 = 64
            g1[3] = (64 << 16);                          // tile_dim0  = 64
            g1[4] = 64;                                  // tile_dim1  = 64
            g1[5] = 64;                                  // tensor_dim0_stride = 64
            g1[6] = 0;
            g1[7] = 0;
            tdm_i32x4 z4 = {0, 0, 0, 0};
#if __clang_major__ >= 23
            tdm_i32x8 z8 = {0, 0, 0, 0, 0, 0, 0, 0};
            __builtin_amdgcn_tensor_load_to_lds(g0, g1, z4, z4, z8, 0);
#else
            __builtin_amdgcn_tensor_load_to_lds(g0, g1, z4, z4, 0);
#endif
        }
#else
        *(v8h*)(kd + tid * 16)     = *(const v8h*)(kg + tid * 16);
        *(v8h*)(kd + tid * 16 + 8) = *(const v8h*)(kg + tid * 16 + 8);
#endif
        // V block, stored transposed so PV B-fragments are contiguous b128.
        const _Float16* vg = vp + ((long)b * S_ + blk * 64) * 64;
        _Float16* vd = Vt + bf * 4096;
        #pragma unroll
        for (int c = 0; c < 2; ++c) {
            const int p   = tid + c * 256;   // 0..511
            const int tok = p & 63;
            const int fb  = (p >> 6) * 8;
            v8h vv = *(const v8h*)(vg + tok * 64 + fb);
            #pragma unroll
            for (int f = 0; f < 8; ++f) vd[(fb + f) * 64 + tok] = vv[f];
        }
    };

    // ---- Preload Q A-fragments (ISA 16-bit A layout, 16x32 per k-slice) ----
    v16h aq[2];
    {
        const _Float16* qp = qwd + (qrow0 + nn) * 64;
        #pragma unroll
        for (int c = 0; c < 2; ++c) {
            v8h lo = *(const v8h*)(qp + c * 32 + hi * 8);
            v8h hh = *(const v8h*)(qp + c * 32 + 16 + hi * 8);
            #pragma unroll
            for (int i = 0; i < 8; ++i) { aq[c][i] = lo[i]; aq[c][i + 8] = hh[i]; }
        }
    }

    v8f oacc[4] = {};
    float mrow[8], lrow[8];
    #pragma unroll
    for (int r = 0; r < 8; ++r) { mrow[r] = -1e30f; lrow[r] = 0.0f; }

    stage(0, 0);   // prologue: stage first block into buffer 0

    for (int i = 0; i < NBLK; ++i) {
        const int buf = i & 1;

        // Publish block i: wave0 fences the TDM; barrier publishes V ds-stores.
#ifdef USE_TDM
        if (wv == 0) __builtin_amdgcn_s_wait_tensorcnt(0);
#endif
        __syncthreads();

        // Kick off staging of block i+1 into the other buffer (overlapped).
        if (i + 1 < NBLK) stage(i + 1, buf ^ 1);

        // Prefetch block i+2 lines into cache (global_prefetch_b8).
        {
            const int kn = ((i + 2) * 64) & (S_ - 1);
            __builtin_prefetch(kwd + ((long)b * S_ + kn) * 64 + tid * 16, 0, 0);
            __builtin_prefetch(vp  + ((long)b * S_ + kn) * 64 + tid * 16, 0, 0);
        }

        const _Float16* kb = Klds + buf * 4096;
        const _Float16* vb = Vt + buf * 4096;

        // ---- Scores: S(16x64) = Q(16x64) . K^T, four 16-token N-tiles ----
        v8f sc[4];
        #pragma unroll
        for (int j = 0; j < 4; ++j) {
            v8f acc = {};
            #pragma unroll
            for (int c = 0; c < 2; ++c) {
                const _Float16* kp = kb + (j * 16 + nn) * 64 + c * 32 + hi * 8;
                v8h lo = *(const v8h*)(kp);
                v8h hh = *(const v8h*)(kp + 16);
                v16h bfrag;
                #pragma unroll
                for (int x = 0; x < 8; ++x) { bfrag[x] = lo[x]; bfrag[x + 8] = hh[x]; }
                acc = __builtin_amdgcn_wmma_f32_16x16x32_f16(
                        false, aq[c], false, bfrag, (short)0, acc, false, false);
            }
            #pragma unroll
            for (int x = 0; x < 8; ++x) acc[x] += mval;
            sc[j] = acc;
        }

        // ---- Online softmax over 64 columns ----
        float pnew[4][8];
        #pragma unroll
        for (int r = 0; r < 8; ++r) {
            float mx = fmaxf(fmaxf(sc[0][r], sc[1][r]), fmaxf(sc[2][r], sc[3][r]));
            #pragma unroll
            for (int d = 8; d >= 1; d >>= 1)
                mx = fmaxf(mx, __shfl_xor(mx, d, 32));
            const float mn   = fmaxf(mrow[r], mx);
            const float corr = __expf(mrow[r] - mn);
            float ps = 0.0f;
            #pragma unroll
            for (int j = 0; j < 4; ++j) {
                pnew[j][r] = __expf(sc[j][r] - mn);
                ps += pnew[j][r];
            }
            #pragma unroll
            for (int d = 8; d >= 1; d >>= 1)
                ps += __shfl_xor(ps, d, 32);
            lrow[r] = lrow[r] * corr + ps;
            mrow[r] = mn;
            #pragma unroll
            for (int t = 0; t < 4; ++t) oacc[t][r] *= corr;
        }

        // ---- Re-layout P (C-format -> A-format) via per-wave LDS ----
        _Float16* pw = Pl + wv * (16 * 64);
        #pragma unroll
        for (int j = 0; j < 4; ++j)
            #pragma unroll
            for (int r = 0; r < 8; ++r)
                pw[(r + 8 * hi) * 64 + j * 16 + nn] = (_Float16)pnew[j][r];

        v16h ap[2];
        #pragma unroll
        for (int c = 0; c < 2; ++c) {
            const _Float16* pp = pw + nn * 64 + c * 32 + hi * 8;
            v8h lo = *(const v8h*)(pp);
            v8h hh = *(const v8h*)(pp + 16);
            #pragma unroll
            for (int x = 0; x < 8; ++x) { ap[c][x] = lo[x]; ap[c][x + 8] = hh[x]; }
        }

        // ---- O += P(16x64) . V(64x64): 4 feature tiles x 2 token k-slices ----
        #pragma unroll
        for (int t = 0; t < 4; ++t) {
            #pragma unroll
            for (int c = 0; c < 2; ++c) {
                const _Float16* vbp = vb + (t * 16 + nn) * 64 + c * 32 + hi * 8;
                v8h lo = *(const v8h*)(vbp);
                v8h hh = *(const v8h*)(vbp + 16);
                v16h vfrag;
                #pragma unroll
                for (int x = 0; x < 8; ++x) { vfrag[x] = lo[x]; vfrag[x + 8] = hh[x]; }
                oacc[t] = __builtin_amdgcn_wmma_f32_16x16x32_f16(
                            false, ap[c], false, vfrag, (short)0, oacc[t], false, false);
            }
        }
    }

    // ---- Epilogue: O / l ----
    #pragma unroll
    for (int t = 0; t < 4; ++t) {
        #pragma unroll
        for (int r = 0; r < 8; ++r) {
            const long row = qrow0 + r + 8 * hi;
            out[row * 64 + t * 16 + nn] = oacc[t][r] / lrow[r];
        }
    }
}

// ---------------------------------------------------------------------------
extern "C" void kernel_launch(void* const* d_in, const int* in_sizes, int n_in,
                              void* d_out, int out_size, void* d_ws, size_t ws_size,
                              hipStream_t stream) {
    const float* q    = (const float*)d_in[0];
    const float* k    = (const float*)d_in[1];
    const float* v    = (const float*)d_in[2];
    const float* mask = (const float*)d_in[3];
    const float* Wq   = (const float*)d_in[4];
    const float* bq   = (const float*)d_in[5];
    const float* Wk   = (const float*)d_in[6];
    const float* bk   = (const float*)d_in[7];
    const float* Wv   = (const float*)d_in[8];
    const float* bv   = (const float*)d_in[9];
    const float* qw   = (const float*)d_in[10];
    const float* kw   = (const float*)d_in[11];
    float* out = (float*)d_out;

    const long N = (long)B_ * S_ * D_;
    _Float16* qwd = (_Float16*)d_ws;
    _Float16* kwd = qwd + N;
    _Float16* vpf = kwd + N;

    proj_kernel<<<(B_ * S_) / 4, 256, 0, stream>>>(
        q, k, v, Wq, bq, Wk, bk, Wv, bv, qw, kw, qwd, kwd, vpf);

    dim3 grid(S_ / 128, B_);
    fa_kernel<<<grid, 256, 0, stream>>>(qwd, kwd, vpf, mask, out);
}
